// GINNet_7756710937226
// MI455X (gfx1250) — compile-verified, matrix operand order checked by reading, not run on hardware
//
#include <hip/hip_runtime.h>

#define N_NODES   100000
#define N_EDGES   640000
#define DIM       128
#define NUM_GRAPHS 64
#define OUT_COLS  (3 * DIM)

typedef __attribute__((ext_vector_type(16))) __bf16 v16bf;
typedef __attribute__((ext_vector_type(8)))  float  v8f;

// ---------- bf16 helpers (bit-level, round-to-nearest-even) ----------
__device__ __forceinline__ unsigned short f32_to_bf16_rne(float f) {
    unsigned int u = __float_as_uint(f);
    unsigned int r = u + 0x7FFFu + ((u >> 16) & 1u);
    return (unsigned short)(r >> 16);
}
__device__ __forceinline__ float bf16_to_f32(unsigned short h) {
    return __uint_as_float(((unsigned int)h) << 16);
}

// ---------- zero the output (pooling accumulates with atomics) ----------
__global__ void zero_kernel(float* __restrict__ out, int n) {
    int i = blockIdx.x * blockDim.x + threadIdx.x;
    if (i < n) out[i] = 0.0f;
}

// ---------- agg = x (the (1+eps)*x_i term, eps = 0) ----------
__global__ void copy_kernel(const float* __restrict__ s, float* __restrict__ d, int n4) {
    int i = blockIdx.x * blockDim.x + threadIdx.x;
    if (i < n4) ((float4*)d)[i] = ((const float4*)s)[i];
}

// ---------- agg[dst] += x[src] : one wave per edge, float4 per lane ----------
__global__ void edge_agg_kernel(const float* __restrict__ x,
                                const int* __restrict__ src,
                                const int* __restrict__ dst,
                                float* __restrict__ agg) {
    int tid  = blockIdx.x * blockDim.x + threadIdx.x;
    int e    = tid >> 5;
    int lane = tid & 31;
    if (e >= N_EDGES) return;
    int s = src[e];
    int d = dst[e];
    float4 v = *(const float4*)(x + (size_t)s * DIM + lane * 4);
    float* o = agg + (size_t)d * DIM + lane * 4;
    atomicAdd(o + 0, v.x);
    atomicAdd(o + 1, v.y);
    atomicAdd(o + 2, v.z);
    atomicAdd(o + 3, v.w);
}

// ---------- f32 -> (bf16 hi, bf16 lo) split, row-major ----------
__global__ void convert_split_kernel(const float* __restrict__ src,
                                     unsigned short* __restrict__ hi,
                                     unsigned short* __restrict__ lo, int n4) {
    int i = blockIdx.x * blockDim.x + threadIdx.x;
    if (i >= n4) return;
    float4 v = ((const float4*)src)[i];
    ushort4 hv, lv;
    hv.x = f32_to_bf16_rne(v.x); lv.x = f32_to_bf16_rne(v.x - bf16_to_f32(hv.x));
    hv.y = f32_to_bf16_rne(v.y); lv.y = f32_to_bf16_rne(v.y - bf16_to_f32(hv.y));
    hv.z = f32_to_bf16_rne(v.z); lv.z = f32_to_bf16_rne(v.z - bf16_to_f32(hv.z));
    hv.w = f32_to_bf16_rne(v.w); lv.w = f32_to_bf16_rne(v.w - bf16_to_f32(hv.w));
    ((ushort4*)hi)[i] = hv;
    ((ushort4*)lo)[i] = lv;
}

// ---------- W[k][n] (f32) -> WT_hi[n][k], WT_lo[n][k] (bf16) ----------
__global__ void weight_split_t_kernel(const float* __restrict__ W,
                                      unsigned short* __restrict__ hiT,
                                      unsigned short* __restrict__ loT) {
    int idx = blockIdx.x * blockDim.x + threadIdx.x;   // 128*128
    int n = idx >> 7;
    int k = idx & 127;
    float w = W[k * DIM + n];
    unsigned short h = f32_to_bf16_rne(w);
    hiT[n * DIM + k] = h;
    loT[n * DIM + k] = f32_to_bf16_rne(w - bf16_to_f32(h));
}

// ---------- out = relu(A @ W^T_rowmajor + bias), split-bf16 WMMA ----------
// Block: 256 threads = 8 waves; block -> 16-row stripe; wave w -> N-tile w.
// A-fragment (16x32 bf16): lane m    holds row m, K = k0+0..7  and k0+16..23
//                          lane m+16 holds row m, K = k0+8..15 and k0+24..31
// B-fragment (32x16 bf16): identical pattern on WT rows (WT[n][k]).
// Both are two contiguous 16B loads per lane.
__global__ void __launch_bounds__(256)
gemm_bias_relu_wmma(const unsigned short* __restrict__ Ahi,
                    const unsigned short* __restrict__ Alo,
                    const unsigned short* __restrict__ BhiT,
                    const unsigned short* __restrict__ BloT,
                    const float* __restrict__ bias,
                    float* __restrict__ out) {
    const int lane = threadIdx.x & 31;
    const int wave = threadIdx.x >> 5;        // 0..7 -> N tile
    const int half = lane >> 4;               // 0 or 1
    const int m    = lane & 15;
    const int rowBase = blockIdx.x * 16;
    const int row  = rowBase + m;             // A row served by this lane
    const int nCol = wave * 16 + m;           // C/D column held by this lane

    const unsigned short* aHiRow = Ahi  + (size_t)row  * DIM;
    const unsigned short* aLoRow = Alo  + (size_t)row  * DIM;
    const unsigned short* bHiRow = BhiT + (size_t)nCol * DIM;
    const unsigned short* bLoRow = BloT + (size_t)nCol * DIM;

    union Frag { uint4 u[2]; v16bf v; };

    v8f acc = {};
#pragma unroll
    for (int k0 = 0; k0 < DIM; k0 += 32) {
        const int o = k0 + half * 8;
        Frag ah, al, bh, bl;
        ah.u[0] = *(const uint4*)(aHiRow + o);
        ah.u[1] = *(const uint4*)(aHiRow + o + 16);
        al.u[0] = *(const uint4*)(aLoRow + o);
        al.u[1] = *(const uint4*)(aLoRow + o + 16);
        bh.u[0] = *(const uint4*)(bHiRow + o);
        bh.u[1] = *(const uint4*)(bHiRow + o + 16);
        bl.u[0] = *(const uint4*)(bLoRow + o);
        bl.u[1] = *(const uint4*)(bLoRow + o + 16);
        // split-fp32: A*B ~= Ahi*Bhi + Ahi*Blo + Alo*Bhi
        acc = __builtin_amdgcn_wmma_f32_16x16x32_bf16(false, ah.v, false, bh.v,
                                                      (short)0, acc, false, false);
        acc = __builtin_amdgcn_wmma_f32_16x16x32_bf16(false, ah.v, false, bl.v,
                                                      (short)0, acc, false, false);
        acc = __builtin_amdgcn_wmma_f32_16x16x32_bf16(false, al.v, false, bh.v,
                                                      (short)0, acc, false, false);
    }

    const float bn = bias[nCol];
#pragma unroll
    for (int r = 0; r < 8; ++r) {
        float v = acc[r] + bn;
        v = v > 0.0f ? v : 0.0f;
        int orow = rowBase + half * 8 + r;     // C/D: lanes 16-31 hold M = r+8
        out[(size_t)orow * DIM + nCol] = v;
    }
}

// ---------- pooled[g] += sum over nodes in g (batch is sorted) ----------
__global__ void pool_kernel(const float* __restrict__ h,
                            const int* __restrict__ batch,
                            float* __restrict__ out, int colOff) {
    int col   = threadIdx.x;                  // 0..127
    int node0 = blockIdx.x * 128;
    float acc = 0.0f;
    int gPrev = batch[node0];
    for (int i = 0; i < 128; ++i) {
        int node = node0 + i;
        if (node >= N_NODES) break;
        int g = batch[node];
        if (g != gPrev) {
            atomicAdd(&out[gPrev * OUT_COLS + colOff + col], acc);
            acc = 0.0f;
            gPrev = g;
        }
        acc += h[(size_t)node * DIM + col];
    }
    atomicAdd(&out[gPrev * OUT_COLS + colOff + col], acc);
}

extern "C" void kernel_launch(void* const* d_in, const int* in_sizes, int n_in,
                              void* d_out, int out_size, void* d_ws, size_t ws_size,
                              hipStream_t stream) {
    const float* x     = (const float*)d_in[0];
    const int*   ei    = (const int*)d_in[1];
    const int*   batch = (const int*)d_in[2];
    const float* w1a   = (const float*)d_in[3];
    const float* b1a   = (const float*)d_in[4];
    const float* w1b   = (const float*)d_in[5];
    const float* b1b   = (const float*)d_in[6];
    const float* w2a   = (const float*)d_in[7];
    const float* b2a   = (const float*)d_in[8];
    const float* w2b   = (const float*)d_in[9];
    const float* b2b   = (const float*)d_in[10];
    float* out = (float*)d_out;

    const int* src = ei;
    const int* dst = ei + N_EDGES;

    // workspace carve-up (~205 MB)
    const size_t NF = (size_t)N_NODES * DIM;        // 12.8M elements
    char* ws = (char*)d_ws;
    float* agg = (float*)ws;                 ws += NF * sizeof(float);
    float* tmp = (float*)ws;                 ws += NF * sizeof(float);
    float* h   = (float*)ws;                 ws += NF * sizeof(float);
    unsigned short* aHi = (unsigned short*)ws; ws += NF * sizeof(unsigned short);
    unsigned short* aLo = (unsigned short*)ws; ws += NF * sizeof(unsigned short);
    unsigned short* wHi = (unsigned short*)ws; ws += 4 * DIM * DIM * sizeof(unsigned short);
    unsigned short* wLo = (unsigned short*)ws;

    const int N4          = (int)(NF / 4);          // 3.2M float4s
    const int CVT_BLOCKS  = N4 / 256;               // 12500
    const int EDGE_BLOCKS = (N_EDGES * 32) / 256;   // 80000
    const int GEMM_BLOCKS = N_NODES / 16;           // 6250
    const int POOL_BLOCKS = (N_NODES + 127) / 128;  // 782

    zero_kernel<<<(out_size + 255) / 256, 256, 0, stream>>>(out, out_size);

    // pre-split + transpose the 4 weight matrices (bf16 hi/lo)
    const float* Wsrc[4] = {w1a, w1b, w2a, w2b};
    for (int i = 0; i < 4; ++i) {
        weight_split_t_kernel<<<64, 256, 0, stream>>>(
            Wsrc[i], wHi + (size_t)i * DIM * DIM, wLo + (size_t)i * DIM * DIM);
    }

    auto run_layer = [&](const float* xin, int wa, const float* ba,
                         int wb, const float* bb, int colOff) {
        copy_kernel<<<CVT_BLOCKS, 256, 0, stream>>>(xin, agg, N4);
        edge_agg_kernel<<<EDGE_BLOCKS, 256, 0, stream>>>(xin, src, dst, agg);
        convert_split_kernel<<<CVT_BLOCKS, 256, 0, stream>>>(agg, aHi, aLo, N4);
        gemm_bias_relu_wmma<<<GEMM_BLOCKS, 256, 0, stream>>>(
            aHi, aLo, wHi + (size_t)wa * DIM * DIM, wLo + (size_t)wa * DIM * DIM, ba, tmp);
        convert_split_kernel<<<CVT_BLOCKS, 256, 0, stream>>>(tmp, aHi, aLo, N4);
        gemm_bias_relu_wmma<<<GEMM_BLOCKS, 256, 0, stream>>>(
            aHi, aLo, wHi + (size_t)wb * DIM * DIM, wLo + (size_t)wb * DIM * DIM, bb, h);
        pool_kernel<<<POOL_BLOCKS, 128, 0, stream>>>(h, batch, out, colOff);
    };

    run_layer(x, 0, b1a, 1, b1b, 0);          // conv1 (nn1)
    run_layer(h, 2, b2a, 3, b2b, DIM);        // conv2 (nn2)
    run_layer(h, 2, b2a, 3, b2b, 2 * DIM);    // conv3 (shares nn2)
}
